// CNActorController_60189671686831
// MI455X (gfx1250) — matrix operand order repaired and checked by reading, not compiled, not run on hardware
//
#include <hip/hip_runtime.h>
#include <hip/hip_bf16.h>
#include <math.h>

// CNActorController for MI455X (gfx1250): f16 WMMA GEMM pipeline, f32 accumulate.
// A-operand staged through LDS by the Tensor Data Mover (tensor_load_to_lds),
// W-operand transposed into LDS cooperatively; 8 v_wmma per K=32 step per wave.
// Shapes: B=8192, A=16, D=64, H=256, C=2, NA=5.  N = B*A = 131072 rows.

typedef __attribute__((ext_vector_type(16))) _Float16 v16h;
typedef __attribute__((ext_vector_type(8)))  _Float16 v8h;
typedef __attribute__((ext_vector_type(8)))  float    v8f;
typedef __attribute__((ext_vector_type(4)))  unsigned int v4u;
typedef __attribute__((ext_vector_type(8)))  int      v8i;
typedef __attribute__((ext_vector_type(4)))  int      v4i;

// Toolchain probe result (round 5): clang-23 / therock-10.0 -> 6-arg
// tensor_load_to_lds: (uint32x4, int32x8, int32x4, int32x4, int32x8, i32 cpol).
// ROCm 7.2 / clang-22 uses the 5-arg form.
#if __has_builtin(__builtin_amdgcn_tensor_load_to_lds) && \
    __has_builtin(__builtin_amdgcn_s_wait_tensorcnt)
#define USE_TDM 1
#else
#define USE_TDM 0
#endif

// Compile-time probe: surfaces in hipcc stderr.
#define PROBE_STR2(x) #x
#define PROBE_STR(x) PROBE_STR2(x)
#pragma message("CDNA5-PROBE clang_major=" PROBE_STR(__clang_major__) " USE_TDM=" PROBE_STR(USE_TDM))

#define DEVINL __device__ __forceinline__

DEVINL float sigm(float x) { return 1.0f / (1.0f + __expf(-x)); }

// ---------------------------------------------------------------- f32 -> f16
__global__ void cvt_f32_f16_kernel(const float* __restrict__ src,
                                   _Float16* __restrict__ dst, long n) {
  long i = (long)blockIdx.x * blockDim.x + threadIdx.x;
  if (i < n) dst[i] = (_Float16)src[i];
}

// ------------------------------------------------------------ WMMA GEMM tile
// A: (N x K) f16 row-major activations, W: (K x M) f16 row-major weights,
// bias: (M) f32.  Out = act(A@W + bias) to optional f32 and/or f16 buffers.
//
// Workgroup tile: 128 rows x 128 cols.  8 waves laid out 2 (row) x 4 (col);
// each wave register-blocks a 64x32 tile = 4x2 WMMA 16x16 tiles.
// LDS: A slab 128 x (K+16) halves (TDM-loaded, padded via D# pad fields),
//      W slab 128 cols x (K+16) halves (cooperative transposed fill).
// Requires N%128==0, K%32==0 (and K power-of-two here), M%128==0.
template <int ACT>
__global__ void __launch_bounds__(256)
gemm_wmma_kernel(const _Float16* __restrict__ A,
                 const _Float16* __restrict__ W,
                 const float* __restrict__ bias,
                 float* __restrict__ outF,
                 _Float16* __restrict__ outH,
                 int N, int K, int M) {
  extern __shared__ _Float16 smem[];
  const int KP = K + 16;                      // padded row pitch (halves)
  _Float16* Al = smem;                        // [128][KP]  A slab
  _Float16* Wl = smem + 128 * KP;             // [128][KP]  W slab (transposed)

  const int tid = threadIdx.x;
  const int n0  = blockIdx.y * 128;           // column block
  const int r0  = blockIdx.x * 128;           // row block
  const int wid = tid >> 5;

#if USE_TDM
  if (wid == 0) {
    // ---- Tensor DMA: A[r0 .. r0+127][0..K) -> Al with K+16-half row pitch.
    // D# group0: [1:0]=count=1, [63:32]=lds_addr, [120:64]=global_addr,
    //            [127:126]=type=2.
    // D# group1: data_size=1(2B); tensor_dim0=K, tensor_dim1=128;
    //            tile_dim0=K, tile_dim1=128; tensor_dim0_stride=K;
    //            pad_enable, pad_interval=K/2 dwords, pad_amount=8 dwords.
    const unsigned long long ga = (unsigned long long)(uintptr_t)(A + (size_t)r0 * K);
    const unsigned int lds = (unsigned int)(uintptr_t)(void*)Al;  // addr[31:0]
    v4u g0 = { 1u, lds, (unsigned int)ga,
               (unsigned int)((ga >> 32) & 0x01FFFFFFu) | (2u << 30) };
    const unsigned int padi = (unsigned int)(__builtin_ctz((unsigned)K) - 2); // 2^(v+1) dw
    const unsigned int Ku = (unsigned int)K & 0xFFFFu;
    v8i g1 = { (int)((1u << 16) | (1u << 20) | (padi << 22) | (7u << 25)),
               (int)(Ku << 16),          // tensor_dim0[15:0]
               (int)(128u << 16),        // tensor_dim1[15:0]=128
               (int)(Ku << 16),          // tile_dim0=K
               (int)128,                 // tile_dim1=128, tile_dim2=0
               (int)K,                   // tensor_dim0_stride[31:0]
               0, 0 };
    v4i g2 = { 0, 0, 0, 0 };
    v4i g3 = { 0, 0, 0, 0 };
#if __clang_major__ >= 23
    v8i g4 = { 0, 0, 0, 0, 0, 0, 0, 0 };   // VADDR4 slot: unused -> zeros
    __builtin_amdgcn_tensor_load_to_lds(g0, g1, g2, g3, g4, 0);
#else
    __builtin_amdgcn_tensor_load_to_lds(g0, g1, g2, g3, 0);
#endif
  }
#else
  {
    // Fallback: cooperative A slab copy with the same padded pitch.
    const int kshift = __builtin_ctz((unsigned)K);
    for (int idx = tid; idx < 128 * K; idx += 256) {
      int row = idx >> kshift;
      int k   = idx & (K - 1);
      Al[row * KP + k] = A[(size_t)(r0 + row) * K + k];
    }
  }
#endif

  // Cooperative transposed fill of the 128-column weight slab (L2-resident).
  for (int idx = tid; idx < 128 * K; idx += 256) {
    int k = idx >> 7;
    int n = idx & 127;
    Wl[n * KP + k] = W[(size_t)k * M + n0 + n];
  }

#if USE_TDM
  if (wid == 0) __builtin_amdgcn_s_wait_tensorcnt(0);
#endif
  __syncthreads();

  const int lane = tid & 31;
  const int wr   = wid >> 2;                  // 0..1  wave row
  const int wc   = wid & 3;                   // 0..3  wave col
  const int mlo  = lane & 15;
  const int hi   = lane >> 4;                 // lane group 0/1

  // A-operand k order used here (contiguous, same as B): lanes 0-15 hold K=kb..kb+15
  // with kb=0; lanes 16-31 same with kb=16.
  const _Float16* Ar[4];
#pragma unroll
  for (int t = 0; t < 4; ++t)
    Ar[t] = Al + (size_t)(wr * 64 + t * 16 + mlo) * KP + hi * 16;
  // B-operand k order used here (contiguous, same as A): lanes 0-15 hold K=0..15, lanes 16-31 K=16..31,
  // N = lane&15.  Contiguous in the transposed LDS slab.
  const _Float16* Br[2];
#pragma unroll
  for (int t = 0; t < 2; ++t)
    Br[t] = Wl + (size_t)(wc * 32 + t * 16 + mlo) * KP + hi * 16;

  v8f acc[4][2] = {};
  for (int k0 = 0; k0 < K; k0 += 32) {
    v16h a[4], b[2];
#pragma unroll
    for (int t = 0; t < 4; ++t) {
      v8h alo = *(const v8h*)(Ar[t] + k0);        // K = k0+kb .. +7
      v8h ahi = *(const v8h*)(Ar[t] + k0 + 8);    // K = k0+kb+8 .. +15
#pragma unroll
      for (int i = 0; i < 8; ++i) { a[t][i] = alo[i]; a[t][8 + i] = ahi[i]; }
    }
#pragma unroll
    for (int t = 0; t < 2; ++t)
      b[t] = *(const v16h*)(Br[t] + k0);          // 32B LDS (2x ds_load_b128)
#pragma unroll
    for (int i = 0; i < 4; ++i)
#pragma unroll
      for (int j = 0; j < 2; ++j)
        acc[i][j] = __builtin_amdgcn_wmma_f32_16x16x32_f16(
            false, a[i], false, b[j], (short)0, acc[i][j], false, false);
  }

  // D layout: VGPR r <-> M = r + 8*hi, N = lane&15.
#pragma unroll
  for (int j = 0; j < 2; ++j) {
    const int col = n0 + wc * 32 + j * 16 + mlo;
    const float bv = bias ? bias[col] : 0.0f;
#pragma unroll
    for (int i = 0; i < 4; ++i) {
#pragma unroll
      for (int r = 0; r < 8; ++r) {
        const int row = r0 + wr * 64 + i * 16 + r + hi * 8;
        float v = acc[i][j][r] + bv;
        if (ACT) v = fmaxf(v, 0.0f);
        if (outF) outF[(size_t)row * M + col] = v;
        if (outH) outH[(size_t)row * M + col] = (_Float16)v;
      }
    }
  }
}

// ------------------------------------------- agent mean + OU noise -> ci_n
// ci_n[b,h] = (1/A) * sum_a (x[b,a,h] + mask[a]*s1[a,h]*s0[a,h])
//             + mask[0]*s3[h]*s2[h]          (identical across agents)
__global__ void aggregate_kernel(const float* __restrict__ xf,
                                 const float* __restrict__ mask,
                                 const float* __restrict__ s0,
                                 const float* __restrict__ s1,
                                 const float* __restrict__ s2,
                                 const float* __restrict__ s3,
                                 _Float16* __restrict__ cin, int Bn) {
  const long idx = (long)blockIdx.x * blockDim.x + threadIdx.x;  // b*256 + h
  const int b = (int)(idx >> 8);
  const int h = (int)(idx & 255);
  if (b >= Bn) return;
  const float* xr = xf + ((size_t)b * 16) * 256 + h;
  float s = 0.0f;
#pragma unroll
  for (int a = 0; a < 16; ++a)
    s += xr[(size_t)a * 256] + mask[a] * s1[a * 256 + h] * s0[a * 256 + h];
  float ci = s * (1.0f / 16.0f) + mask[0] * s3[h] * s2[h];
  cin[idx] = (_Float16)ci;
}

// -------------------------------------------------- GRU gates + residual
// x_new = z*x + (1-z)*tanh(xh + r*rh) + h0,  z/r sigmoids (reset_after GRU).
__global__ void gru_update_kernel(const float* __restrict__ mx,   // (B,768)
                                  const float* __restrict__ mh,   // (N,768)
                                  const _Float16* __restrict__ h0h,
                                  float* __restrict__ xf,
                                  _Float16* __restrict__ xh,
                                  long Nrow) {
  const long idx = (long)blockIdx.x * blockDim.x + threadIdx.x;  // n*256 + h
  const long n = idx >> 8;
  const int  h = (int)(idx & 255);
  if (n >= Nrow) return;
  const long b = n >> 4;
  const float xz = mx[b * 768 + h];
  const float xr = mx[b * 768 + 256 + h];
  const float xn = mx[b * 768 + 512 + h];
  const float rz = mh[n * 768 + h];
  const float rr = mh[n * 768 + 256 + h];
  const float rh = mh[n * 768 + 512 + h];
  const float z  = sigm(xz + rz);
  const float r  = sigm(xr + rr);
  const float hh = tanhf(xn + r * rh);
  const float xo = xf[idx];
  const float xnew = z * xo + (1.0f - z) * hh + (float)h0h[idx];
  xf[idx] = xnew;
  xh[idx] = (_Float16)xnew;
}

// ------------------------------------------------- final (256 -> 5) logits
__global__ void out_proj_kernel(const float* __restrict__ dec,
                                const float* __restrict__ Wout,
                                const float* __restrict__ bout,
                                float* __restrict__ out, long Nrow) {
  const long n = (long)blockIdx.x * blockDim.x + threadIdx.x;
  if (n >= Nrow) return;
  float acc[5];
#pragma unroll
  for (int j = 0; j < 5; ++j) acc[j] = bout[j];
  const float* d = dec + n * 256;
  for (int k = 0; k < 256; ++k) {
    const float dv = d[k];
#pragma unroll
    for (int j = 0; j < 5; ++j) acc[j] = fmaf(dv, Wout[k * 5 + j], acc[j]);
  }
#pragma unroll
  for (int j = 0; j < 5; ++j) out[n * 5 + j] = acc[j];
}

// ---------------------------------------------------------------------------
static void launch_gemm(const _Float16* A, const _Float16* W, const float* bias,
                        float* outF, _Float16* outH, int N, int K, int M,
                        bool relu, hipStream_t s) {
  dim3 grid(N / 128, M / 128);
  size_t lds = (size_t)2 * 128 * (K + 16) * sizeof(_Float16);  // A slab + W slab
  if (relu)
    gemm_wmma_kernel<1><<<grid, 256, lds, s>>>(A, W, bias, outF, outH, N, K, M);
  else
    gemm_wmma_kernel<0><<<grid, 256, lds, s>>>(A, W, bias, outF, outH, N, K, M);
}

static void launch_cvt(const float* src, _Float16* dst, long n, hipStream_t s) {
  cvt_f32_f16_kernel<<<(unsigned)((n + 255) / 256), 256, 0, s>>>(src, dst, n);
}

extern "C" void kernel_launch(void* const* d_in, const int* in_sizes, int n_in,
                              void* d_out, int out_size, void* d_ws, size_t ws_size,
                              hipStream_t stream) {
  (void)in_sizes; (void)n_in; (void)out_size; (void)ws_size;
  const float* obs    = (const float*)d_in[0];
  const float* mask   = (const float*)d_in[1];
  const float* ou_s0  = (const float*)d_in[2];   // (C,1,A,H)
  const float* ou_s1  = (const float*)d_in[3];   // (C,1,A,H)
  const float* ou_s2  = (const float*)d_in[4];   // (C,1,1,H)
  const float* ou_s3  = (const float*)d_in[5];   // (C,1,1,H)
  const float* W_enc  = (const float*)d_in[6];   // (64,256)
  const float* b_enc  = (const float*)d_in[7];
  const float* W_enc2 = (const float*)d_in[8];   // (256,256)
  const float* b_enc2 = (const float*)d_in[9];
  const float* gru_k  = (const float*)d_in[10];  // (C,256,768)
  const float* gru_rk = (const float*)d_in[11];  // (C,256,768)
  const float* gru_b  = (const float*)d_in[12];  // (C,2,768)
  const float* W_dec  = (const float*)d_in[13];  // (256,256)
  const float* b_dec  = (const float*)d_in[14];
  const float* W_out  = (const float*)d_in[15];  // (256,5)
  const float* b_out  = (const float*)d_in[16];
  float* out = (float*)d_out;

  constexpr int  Bn = 8192, Ag = 16, Dd = 64, Hh = 256, Cc = 2;
  constexpr long Nrow = (long)Bn * Ag;           // 131072

  // ------------------------------------------------ workspace carve-up
  char* ws = (char*)d_ws;
  size_t off = 0;
  auto carve = [&](size_t bytes) -> void* {
    void* p = ws + off;
    off = (off + bytes + 255) & ~(size_t)255;
    return p;
  };
  _Float16* wenc_h  = (_Float16*)carve((size_t)Dd * Hh * 2);
  _Float16* wenc2_h = (_Float16*)carve((size_t)Hh * Hh * 2);
  _Float16* gruk_h  = (_Float16*)carve((size_t)Cc * Hh * 3 * Hh * 2);
  _Float16* grurk_h = (_Float16*)carve((size_t)Cc * Hh * 3 * Hh * 2);
  _Float16* wdec_h  = (_Float16*)carve((size_t)Hh * Hh * 2);
  _Float16* obs_h   = (_Float16*)carve((size_t)Nrow * Dd * 2);
  _Float16* x1_h    = (_Float16*)carve((size_t)Nrow * Hh * 2);
  _Float16* h0_h    = (_Float16*)carve((size_t)Nrow * Hh * 2);
  float*    x_f     = (float*)   carve((size_t)Nrow * Hh * 4);
  _Float16* x_h     = (_Float16*)carve((size_t)Nrow * Hh * 2);
  _Float16* cin_h   = (_Float16*)carve((size_t)Bn * Hh * 2);
  float*    mx_f    = (float*)   carve((size_t)Bn * 3 * Hh * 4);
  float*    mh_f    = (float*)   carve((size_t)Nrow * 3 * Hh * 4);
  float*    dec_f   = mh_f;                      // reuse after GRU layers

  // ------------------------------------------------ f32 -> f16 staging
  launch_cvt(W_enc,  wenc_h,  (long)Dd * Hh, stream);
  launch_cvt(W_enc2, wenc2_h, (long)Hh * Hh, stream);
  launch_cvt(gru_k,  gruk_h,  (long)Cc * Hh * 3 * Hh, stream);
  launch_cvt(gru_rk, grurk_h, (long)Cc * Hh * 3 * Hh, stream);
  launch_cvt(W_dec,  wdec_h,  (long)Hh * Hh, stream);
  launch_cvt(obs,    obs_h,   Nrow * Dd, stream);

  // ------------------------------------------------ encoder stack
  // x1 = relu(obs @ W_enc + b_enc)           (N,64)x(64,256)
  launch_gemm(obs_h, wenc_h, b_enc, nullptr, x1_h, (int)Nrow, Dd, Hh, true, stream);
  // h0 = relu(x1 @ W_enc2 + b_enc2)          (N,256)x(256,256); x_f = h0 (f32)
  launch_gemm(x1_h, wenc2_h, b_enc2, x_f, h0_h, (int)Nrow, Hh, Hh, true, stream);
  // x (f16 GEMM copy) = h0
  (void)hipMemcpyAsync(x_h, h0_h, (size_t)Nrow * Hh * 2, hipMemcpyDeviceToDevice, stream);

  // ------------------------------------------------ C GRU/comm layers
  for (int i = 0; i < Cc; ++i) {
    const float* s0 = ou_s0 + (size_t)i * Ag * Hh;
    const float* s1 = ou_s1 + (size_t)i * Ag * Hh;
    const float* s2 = ou_s2 + (size_t)i * Hh;
    const float* s3 = ou_s3 + (size_t)i * Hh;
    const float* bi = gru_b + (size_t)i * 2 * 3 * Hh;        // input bias
    const float* br = bi + 3 * Hh;                           // recurrent bias

    // ci_n per batch row (identical across agents -> 16x less GEMM work)
    aggregate_kernel<<<(unsigned)(((long)Bn * Hh + 255) / 256), 256, 0, stream>>>(
        x_f, mask, s0, s1, s2, s3, cin_h, Bn);
    // mx = ci_n @ gru_k[i] + gru_b[i,0]      (B,256)x(256,768)
    launch_gemm(cin_h, gruk_h + (size_t)i * Hh * 3 * Hh, bi,
                mx_f, nullptr, Bn, Hh, 3 * Hh, false, stream);
    // mh = x @ gru_rk[i] + gru_b[i,1]        (N,256)x(256,768)
    launch_gemm(x_h, grurk_h + (size_t)i * Hh * 3 * Hh, br,
                mh_f, nullptr, (int)Nrow, Hh, 3 * Hh, false, stream);
    // gates + state update + residual
    gru_update_kernel<<<(unsigned)((Nrow * Hh + 255) / 256), 256, 0, stream>>>(
        mx_f, mh_f, h0_h, x_f, x_h, Nrow);
  }

  // ------------------------------------------------ decoder + logits
  // dec = relu(x @ W_dec + b_dec)            (N,256)x(256,256)
  launch_gemm(x_h, wdec_h, b_dec, dec_f, nullptr, (int)Nrow, Hh, Hh, true, stream);
  // out = dec @ W_out + b_out                (N,256)x(256,5) — scalar kernel
  out_proj_kernel<<<(unsigned)((Nrow + 255) / 256), 256, 0, stream>>>(
      dec_f, W_out, b_out, out, Nrow);
}
